// MultiheadAttention_20658792694509
// MI455X (gfx1250) — matrix operand order repaired and checked by reading, not compiled, not run on hardware
//
#include <hip/hip_runtime.h>

#define B_  4
#define S_  2048
#define D_  1024
#define H_  16
#define DK_ 64
#define KB_ 64
#define NT_ (S_ / KB_)   // 32 kv tiles
#define PAD_ 72          // padded LDS row: 72 halfs = 36 dwords -> conflict-free

typedef _Float16 half_t;
typedef __attribute__((ext_vector_type(16))) _Float16 v16h;
typedef __attribute__((ext_vector_type(16), aligned(16))) _Float16 v16h_a16;
typedef __attribute__((ext_vector_type(8)))  _Float16 v8h;
typedef __attribute__((ext_vector_type(4)))  _Float16 v4h;
typedef __attribute__((ext_vector_type(2)))  _Float16 h2;
typedef __attribute__((ext_vector_type(8)))  float    v8f;

union FragA { v16h v; v8h h[2]; };
union Pack4 { h2 p[2]; v4h v; };
union Pack8 { v4h q[2]; v8h v; };

#define WMMA_F16(a, bm, c) \
    __builtin_amdgcn_wmma_f32_16x16x32_f16(false, (a), false, (bm), (short)0, (c), false, false)

__device__ __forceinline__ h2 pk2(float a, float b) {
    return __builtin_bit_cast(h2, __builtin_amdgcn_cvt_pkrtz(a, b));
}

#if __has_builtin(__builtin_amdgcn_exp2f)
#define EXP2(x) __builtin_amdgcn_exp2f(x)
#else
#define EXP2(x) __expf((x) * 0.69314718055994530942f)
#endif
#if __has_builtin(__builtin_amdgcn_rcpf)
#define RCP(x) __builtin_amdgcn_rcpf(x)
#else
#define RCP(x) (1.0f / (x))
#endif

// async global->LDS DMA (GLOBAL_LOAD_ASYNC_TO_LDS_B128, tracked by ASYNCcnt)
// toolchain signature (from diagnostics): (v4i AS1* src, v4i AS3* ldsdst, imm, imm)
#if __has_builtin(__builtin_amdgcn_global_load_async_to_lds_b128)
#define HAVE_ASYNC_LDS 1
typedef __attribute__((ext_vector_type(4))) int v4i_t;
typedef __attribute__((address_space(1))) v4i_t as1_v4i;
typedef __attribute__((address_space(3))) v4i_t as3_v4i;
#else
#define HAVE_ASYNC_LDS 0
#endif

__device__ __forceinline__ void wait_async0() {
#if __has_builtin(__builtin_amdgcn_s_wait_asynccnt)
    __builtin_amdgcn_s_wait_asynccnt(0);
#else
    asm volatile("s_wait_asynccnt 0x0" ::: "memory");
#endif
}

__device__ __forceinline__ void lds_wave_fence() {
    asm volatile("s_wait_dscnt 0" ::: "memory");
}

#if __has_builtin(__builtin_amdgcn_permlane16)
__device__ __forceinline__ float xor16(float v, unsigned s0, unsigned s1) {
    union { float f; unsigned u; } a, r;
    a.f = v;
    r.u = __builtin_amdgcn_permlane16(a.u, a.u, s0, s1, false, false);
    return r.f;
}
#define ROWMAX_BUTTERFLY(x) do {                                \
    x = fmaxf(x, xor16(x, 0x67452301u, 0xEFCDAB89u)); /*xor1*/  \
    x = fmaxf(x, xor16(x, 0x54761032u, 0xDCFE98BAu)); /*xor2*/  \
    x = fmaxf(x, xor16(x, 0x32107654u, 0xBA98FEDCu)); /*xor4*/  \
    x = fmaxf(x, xor16(x, 0xFEDCBA98u, 0x76543210u)); /*xor8*/  \
} while (0)
#else
#define ROWMAX_BUTTERFLY(x) do {                                \
    x = fmaxf(x, __shfl_xor(x, 1, 32));                         \
    x = fmaxf(x, __shfl_xor(x, 2, 32));                         \
    x = fmaxf(x, __shfl_xor(x, 4, 32));                         \
    x = fmaxf(x, __shfl_xor(x, 8, 32));                         \
} while (0)
#endif

// ---- pre-pass: convert K and V to f16 once (instead of 32x inside the loop) ----
__global__ __launch_bounds__(256)
void cvt_kv_kernel(const float* __restrict__ kf, const float* __restrict__ vf,
                   half_t* __restrict__ kh, half_t* __restrict__ vh)
{
    const size_t idx = ((size_t)blockIdx.x * 256 + threadIdx.x) * 8;
    float4 a = *(const float4*)(kf + idx);
    float4 b = *(const float4*)(kf + idx + 4);
    Pack8 ko;
    Pack4 t0; t0.p[0] = pk2(a.x, a.y); t0.p[1] = pk2(a.z, a.w); ko.q[0] = t0.v;
    Pack4 t1; t1.p[0] = pk2(b.x, b.y); t1.p[1] = pk2(b.z, b.w); ko.q[1] = t1.v;
    *(v8h*)(kh + idx) = ko.v;
    a = *(const float4*)(vf + idx);
    b = *(const float4*)(vf + idx + 4);
    Pack8 vo;
    t0.p[0] = pk2(a.x, a.y); t0.p[1] = pk2(a.z, a.w); vo.q[0] = t0.v;
    t1.p[0] = pk2(b.x, b.y); t1.p[1] = pk2(b.z, b.w); vo.q[1] = t1.v;
    *(v8h*)(vh + idx) = vo.v;
}

template <bool SRC16>
__global__ __launch_bounds__(128)
void mha_fwd_kernel(const float*  __restrict__ q,
                    const float*  __restrict__ kf,
                    const float*  __restrict__ vf,
                    const half_t* __restrict__ kh,
                    const half_t* __restrict__ vh,
                    float* __restrict__ out)
{
    constexpr int   QB     = 64;   // q rows per block (16 per wave)
    constexpr float qscale = 0.125f * 1.44269504088896340736f;  // 1/sqrt(64)*log2(e)

    __shared__ __align__(64) half_t Ks[2][KB_][PAD_];  // 18 KB
    __shared__ __align__(64) half_t Vt[2][DK_][PAD_];  // 18 KB (transposed V)
    __shared__ __align__(64) half_t Ps[4][16][PAD_];   //  9 KB per-wave P scratch

    const int tid  = threadIdx.x;
    const int wave = tid >> 5;
    const int lane = tid & 31;
    const int ln16 = lane & 15;
    const int hsel = lane >> 4;

    const int qblocks = S_ / QB;               // 32
    const int bh = blockIdx.x / qblocks;
    const int qb = blockIdx.x % qblocks;
    const int b  = bh / H_;
    const int h  = bh % H_;

    const size_t base = ((size_t)b * S_) * D_ + (size_t)h * DK_;
    const float*  qbase  = q  + base;
    const float*  kbase  = kf + base;
    const float*  vbase  = vf + base;
    const half_t* khbase = kh + base;
    const half_t* vhbase = vh + base;
    float*        obase  = out + base;

    // cooperative staging: thread owns 32 contiguous elements of one kv row
    const int ldr = tid >> 1;          // kv row 0..63
    const int ldc = (tid & 1) * 32;    // dk col base

    // ---- Q tile -> two 16x32 A fragments (scale folded in) ----
    FragA qf[2];
    {
        const int qrow = qb * QB + wave * 16 + ln16;
        const float* qp = qbase + (size_t)qrow * D_;
        #pragma unroll
        for (int f = 0; f < 2; ++f) {
            #pragma unroll
            for (int e = 0; e < 8; e += 2) {
                h2 lo = pk2(qp[32*f + 16*hsel +     e] * qscale,
                            qp[32*f + 16*hsel +     e + 1] * qscale);
                h2 hi = pk2(qp[32*f + 16*hsel + 8 + e] * qscale,
                            qp[32*f + 16*hsel + 8 + e + 1] * qscale);
                qf[f].v[e]     = lo[0]; qf[f].v[e + 1]     = lo[1];
                qf[f].v[8 + e] = hi[0]; qf[f].v[8 + e + 1] = hi[1];
            }
        }
    }

    v16h vones;
    #pragma unroll
    for (int e = 0; e < 16; ++e) vones[e] = (half_t)1.0f;

    v8f o0 = {}, o1 = {}, o2 = {}, o3 = {};
    float mrow[8], lrow[8];
    #pragma unroll
    for (int r = 0; r < 8; ++r) { mrow[r] = -__builtin_inff(); lrow[r] = 0.0f; }

    // ---- prologue: stage tile 0 into buffer 0 ----
    if constexpr (SRC16) {
        const half_t* kp = khbase + (size_t)ldr * D_ + ldc;
        const half_t* vp = vhbase + (size_t)ldr * D_ + ldc;
        #pragma unroll
        for (int i = 0; i < 4; ++i) {
            v8h k8 = *(const v8h*)(kp + 8*i);
            v8h v8 = *(const v8h*)(vp + 8*i);
            *(v8h*)&Ks[0][ldr][ldc + 8*i] = k8;
            #pragma unroll
            for (int j = 0; j < 8; ++j) Vt[0][ldc + 8*i + j][ldr] = v8[j];
        }
    } else {
        const float* kp = kbase + (size_t)ldr * D_ + ldc;
        const float* vp = vbase + (size_t)ldr * D_ + ldc;
        #pragma unroll
        for (int i = 0; i < 32; i += 4) {
            float4 k4 = *(const float4*)(kp + i);
            float4 v4 = *(const float4*)(vp + i);
            Pack4 kk; kk.p[0] = pk2(k4.x, k4.y); kk.p[1] = pk2(k4.z, k4.w);
            *(v4h*)&Ks[0][ldr][ldc + i] = kk.v;
            h2 va = pk2(v4.x, v4.y), vb = pk2(v4.z, v4.w);
            Vt[0][ldc+i+0][ldr] = va[0];
            Vt[0][ldc+i+1][ldr] = va[1];
            Vt[0][ldc+i+2][ldr] = vb[0];
            Vt[0][ldc+i+3][ldr] = vb[1];
        }
    }
    __syncthreads();

    for (int t = 0; t < NT_; ++t) {
        const int  cur  = t & 1;
        const int  nxt  = cur ^ 1;
        const bool more = (t + 1 < NT_);

        // ---- software pipeline: next tile in flight ----
        float4 kreg32[8], vreg32[8];
        v8h    kreg16[4], vreg16[4];
        if (more) {
            if constexpr (SRC16) {
                const half_t* kp = khbase + (size_t)((t+1)*KB_ + ldr) * D_ + ldc;
                const half_t* vp = vhbase + (size_t)((t+1)*KB_ + ldr) * D_ + ldc;
#if HAVE_ASYNC_LDS
                // K tile: pure byte copy -> async DMA straight into LDS
                #pragma unroll
                for (int i = 0; i < 4; ++i)
                    __builtin_amdgcn_global_load_async_to_lds_b128(
                        (as1_v4i*)(kp + 8*i),
                        (as3_v4i*)&Ks[nxt][ldr][ldc + 8*i], 0, 0);
#else
                #pragma unroll
                for (int i = 0; i < 4; ++i) kreg16[i] = *(const v8h*)(kp + 8*i);
#endif
                #pragma unroll
                for (int i = 0; i < 4; ++i) vreg16[i] = *(const v8h*)(vp + 8*i);
            } else {
                const float* kp = kbase + (size_t)((t+1)*KB_ + ldr) * D_ + ldc;
                const float* vp = vbase + (size_t)((t+1)*KB_ + ldr) * D_ + ldc;
                #pragma unroll
                for (int i = 0; i < 8; ++i) {
                    kreg32[i] = *(const float4*)(kp + 4*i);
                    vreg32[i] = *(const float4*)(vp + 4*i);
                }
            }
        }

        // ---- scores: 4 N-subtiles x 2 K-chunks = 8 WMMAs ----
        v8f s0 = {}, s1 = {}, s2 = {}, s3 = {};
        #pragma unroll
        for (int f = 0; f < 2; ++f) {
            v16h bk0 = *(const v16h_a16*)&Ks[cur][ 0 + ln16][32*f + 16*hsel];
            v16h bk1 = *(const v16h_a16*)&Ks[cur][16 + ln16][32*f + 16*hsel];
            v16h bk2 = *(const v16h_a16*)&Ks[cur][32 + ln16][32*f + 16*hsel];
            v16h bk3 = *(const v16h_a16*)&Ks[cur][48 + ln16][32*f + 16*hsel];
            s0 = WMMA_F16(qf[f].v, bk0, s0);
            s1 = WMMA_F16(qf[f].v, bk1, s1);
            s2 = WMMA_F16(qf[f].v, bk2, s2);
            s3 = WMMA_F16(qf[f].v, bk3, s3);
        }

        // ---- online softmax in exp2 domain (row M = r + 8*hsel) ----
        #pragma unroll
        for (int r = 0; r < 8; ++r) {
            float bm = fmaxf(fmaxf(s0[r], s1[r]), fmaxf(s2[r], s3[r]));
            ROWMAX_BUTTERFLY(bm);
            float mnew = fmaxf(mrow[r], bm);
            float corr = EXP2(mrow[r] - mnew);
            float p0 = EXP2(s0[r] - mnew);
            float p1 = EXP2(s1[r] - mnew);
            float p2 = EXP2(s2[r] - mnew);
            float p3 = EXP2(s3[r] - mnew);
            h2 pa = pk2(p0, p1);
            h2 pb = pk2(p2, p3);
            Ps[wave][r + 8*hsel][ 0 + ln16] = pa[0];
            Ps[wave][r + 8*hsel][16 + ln16] = pa[1];
            Ps[wave][r + 8*hsel][32 + ln16] = pb[0];
            Ps[wave][r + 8*hsel][48 + ln16] = pb[1];
            mrow[r] = mnew;
            lrow[r] *= corr;
            o0[r] *= corr; o1[r] *= corr; o2[r] *= corr; o3[r] *= corr;
        }

        // ---- stage next tile (V transpose-scatter; K unless DMA'd) ----
        if (more) {
            if constexpr (SRC16) {
#if !HAVE_ASYNC_LDS
                #pragma unroll
                for (int i = 0; i < 4; ++i)
                    *(v8h*)&Ks[nxt][ldr][ldc + 8*i] = kreg16[i];
#endif
                #pragma unroll
                for (int i = 0; i < 4; ++i) {
                    #pragma unroll
                    for (int j = 0; j < 8; ++j)
                        Vt[nxt][ldc + 8*i + j][ldr] = vreg16[i][j];
                }
            } else {
                #pragma unroll
                for (int i = 0; i < 8; ++i) {
                    Pack4 kk; kk.p[0] = pk2(kreg32[i].x, kreg32[i].y);
                              kk.p[1] = pk2(kreg32[i].z, kreg32[i].w);
                    *(v4h*)&Ks[nxt][ldr][ldc + 4*i] = kk.v;
                    h2 va = pk2(vreg32[i].x, vreg32[i].y);
                    h2 vb = pk2(vreg32[i].z, vreg32[i].w);
                    Vt[nxt][ldc + 4*i + 0][ldr] = va[0];
                    Vt[nxt][ldc + 4*i + 1][ldr] = va[1];
                    Vt[nxt][ldc + 4*i + 2][ldr] = vb[0];
                    Vt[nxt][ldc + 4*i + 3][ldr] = vb[1];
                }
            }
        }
        lds_wave_fence();

        // ---- P (16x64) back as two A fragments ----
        FragA pf0, pf1;
        pf0.h[0] = *(const v8h*)&Ps[wave][ln16][ 0 + 16*hsel];
        pf0.h[1] = *(const v8h*)&Ps[wave][ln16][ 8 + 16*hsel];
        pf1.h[0] = *(const v8h*)&Ps[wave][ln16][32 + 16*hsel];
        pf1.h[1] = *(const v8h*)&Ps[wave][ln16][40 + 16*hsel];

        // ---- row sums via WMMA against ones ----
        v8f rs = {};
        rs = WMMA_F16(pf0.v, vones, rs);
        rs = WMMA_F16(pf1.v, vones, rs);
        #pragma unroll
        for (int r = 0; r < 8; ++r) lrow[r] += rs[r];

        // ---- O += P * V : 4 N-subtiles x 2 K-chunks = 8 WMMAs ----
        {
            v16h bv;
            bv = *(const v16h_a16*)&Vt[cur][ 0 + ln16][     16*hsel]; o0 = WMMA_F16(pf0.v, bv, o0);
            bv = *(const v16h_a16*)&Vt[cur][ 0 + ln16][32 + 16*hsel]; o0 = WMMA_F16(pf1.v, bv, o0);
            bv = *(const v16h_a16*)&Vt[cur][16 + ln16][     16*hsel]; o1 = WMMA_F16(pf0.v, bv, o1);
            bv = *(const v16h_a16*)&Vt[cur][16 + ln16][32 + 16*hsel]; o1 = WMMA_F16(pf1.v, bv, o1);
            bv = *(const v16h_a16*)&Vt[cur][32 + ln16][     16*hsel]; o2 = WMMA_F16(pf0.v, bv, o2);
            bv = *(const v16h_a16*)&Vt[cur][32 + ln16][32 + 16*hsel]; o2 = WMMA_F16(pf1.v, bv, o2);
            bv = *(const v16h_a16*)&Vt[cur][48 + ln16][     16*hsel]; o3 = WMMA_F16(pf0.v, bv, o3);
            bv = *(const v16h_a16*)&Vt[cur][48 + ln16][32 + 16*hsel]; o3 = WMMA_F16(pf1.v, bv, o3);
        }

#if HAVE_ASYNC_LDS
        if constexpr (SRC16) wait_async0();   // K DMA landed before barrier
#endif
        __syncthreads();   // single barrier per tile (double-buffered)
    }

    // ---- normalize and store fp32 output ----
    #pragma unroll
    for (int r = 0; r < 8; ++r) {
        const float inv = RCP(lrow[r]);
        const int row = qb * QB + wave * 16 + r + 8 * hsel;
        float* op = obase + (size_t)row * D_;
        op[ 0 + ln16] = o0[r] * inv;
        op[16 + ln16] = o1[r] * inv;
        op[32 + ln16] = o2[r] * inv;
        op[48 + ln16] = o3[r] * inv;
    }
}

extern "C" void kernel_launch(void* const* d_in, const int* in_sizes, int n_in,
                              void* d_out, int out_size, void* d_ws, size_t ws_size,
                              hipStream_t stream) {
    (void)in_sizes; (void)n_in; (void)out_size;
    const float* q  = (const float*)d_in[0];
    const float* kf = (const float*)d_in[1];
    const float* vf = (const float*)d_in[2];
    float* out = (float*)d_out;

    const size_t nelem = (size_t)B_ * S_ * D_;            // 8.4M per tensor
    const size_t need  = 2 * nelem * sizeof(half_t);      // 32 MB for Kf16+Vf16
    dim3 grid(B_ * H_ * (S_ / 64));
    dim3 block(128);

    if (d_ws != nullptr && ws_size >= need) {
        half_t* kh = (half_t*)d_ws;
        half_t* vh = kh + nelem;
        cvt_kv_kernel<<<dim3((unsigned)(nelem / (256 * 8))), dim3(256), 0, stream>>>(kf, vf, kh, vh);
        mha_fwd_kernel<true><<<grid, block, 0, stream>>>(q, kf, vf, kh, vh, out);
    } else {
        mha_fwd_kernel<false><<<grid, block, 0, stream>>>(
            q, kf, vf, (const half_t*)kf, (const half_t*)vf, out);
    }
}
